// GCN_21466246545615
// MI455X (gfx1250) — matrix operand order repaired and checked by reading, not compiled, not run on hardware
//
#include <hip/hip_runtime.h>
#include <hip/hip_bf16.h>

#define N_NODES 50000
#define N_EDGES 500000
#define F_INN   6
#define H       146
#define HP      160     // padded hidden (K and N pad for WMMA)
#define GRAPHS  64
#define H2      73
#define H4      36
#define NCLS    10
#define MTILES  (N_NODES / 16)   // 3125 exactly
#define NTILES  (HP / 16)        // 10
#define WAVES_PER_BLOCK 8

typedef __attribute__((ext_vector_type(16))) _Float16 v16h;
typedef __attribute__((ext_vector_type(8)))  _Float16 v8h;
typedef __attribute__((ext_vector_type(8)))  float    v8f;

// ---------------- utility ----------------
__global__ void k_fill(float* p, float v, int n) {
    int i = blockIdx.x * blockDim.x + threadIdx.x;
    if (i < n) p[i] = v;
}

// ---------------- input linear: h = x @ W_in + b_in  (K=6, scalar) ----------------
// Dual-writes f32 feature buffer and the f16 staging buffer for the WMMA GEMM.
__global__ void k_input_linear(const float* __restrict__ x, const float* __restrict__ W,
                               const float* __restrict__ b, float* __restrict__ h,
                               _Float16* __restrict__ h16) {
    int i = blockIdx.x;          // node
    int f = threadIdx.x;         // 0..HP-1
    float v = 0.f;
    if (f < H) {
        float acc = b[f];
#pragma unroll
        for (int k = 0; k < F_INN; ++k) acc += x[i * F_INN + k] * W[k * H + f];
        v = acc;
    }
    h[i * HP + f]   = v;          // zero padding cols
    h16[i * HP + f] = (_Float16)v;
}

// ---------------- degree / norm precompute ----------------
__global__ void k_deg(const int* __restrict__ col, const float* __restrict__ ew, float* deg) {
    int e = blockIdx.x * blockDim.x + threadIdx.x;
    if (e < N_EDGES) atomicAdd(&deg[col[e]], ew[e]);
}

__global__ void k_dinv(const float* __restrict__ deg, float* dinv, float* dinv0, float* selfc) {
    int i = blockIdx.x * blockDim.x + threadIdx.x;
    if (i >= N_NODES) return;
    float d = deg[i];
    dinv[i] = (d > 0.f) ? rsqrtf(fmaxf(d, 1e-30f)) : 0.f;
    float r = rsqrtf(d + 1.f);   // self-loop adds weight 1, always > 0
    dinv0[i] = r;
    selfc[i] = r * r;
}

__global__ void k_norm(const int* __restrict__ row, const int* __restrict__ col,
                       const float* __restrict__ ew,
                       const float* __restrict__ dinv, const float* __restrict__ dinv0,
                       float* normB, float* normA) {
    int e = blockIdx.x * blockDim.x + threadIdx.x;
    if (e >= N_EDGES) return;
    float w = ew[e];
    int r = row[e], c = col[e];
    normA[e] = dinv0[r] * w * dinv0[c];   // layer 0 (with self loops)
    normB[e] = dinv[r]  * w * dinv[c];    // layers 1..3
}

// W[H][H] -> Wt_f16[HP][HP] with Wt[n][k] = W[k][n], zero-padded
__global__ void k_w_to_f16t(const float* __restrict__ W, _Float16* __restrict__ Wt) {
    int idx = blockIdx.x * blockDim.x + threadIdx.x;
    if (idx >= HP * HP) return;
    int n = idx / HP, k = idx % HP;
    float v = (n < H && k < H) ? W[k * H + n] : 0.f;
    Wt[n * HP + k] = (_Float16)v;
}

// ---------------- WMMA GEMM: C[N,HP] = A_f16[N,HP] @ W  (B transposed, LDS-staged) ------
// One wave computes a 16x160 output strip (10 N-tiles): A fragment loaded once per K-step,
// reused across 10 v_wmma ops. B (51 KB) staged once per block into LDS.
__global__ void __launch_bounds__(256)
k_wmma_gemm(const _Float16* __restrict__ A,
            const _Float16* __restrict__ Bt,   // [HP][HP], Bt[n][k]
            float* __restrict__ C) {
    __shared__ v8h sB8[HP * HP / 8];           // 51,200 bytes, 16B-aligned
    _Float16* sB = (_Float16*)sB8;

    // cooperative stage: global_load_b128 -> ds_store_b128
    const v8h* gB = (const v8h*)Bt;
    for (int i = threadIdx.x; i < HP * HP / 8; i += 256) sB8[i] = gB[i];
    __syncthreads();

    const int wave  = threadIdx.x >> 5;
    const int lane  = threadIdx.x & 31;
    const int tileM = blockIdx.x * WAVES_PER_BLOCK + wave;
    if (tileM >= MTILES) return;               // wave-uniform: EXEC stays all-ones
    const int m  = lane & 15;                  // row (A) / col (B,C)
    const int hi = lane >> 4;                  // lane half selects K sub-block
    const _Float16* arow = A + (size_t)(tileM * 16 + m) * HP;

    v8f acc[NTILES];
#pragma unroll
    for (int n = 0; n < NTILES; ++n) acc[n] = (v8f){};

#pragma unroll
    for (int k0 = 0; k0 < HP; k0 += 32) {
        // A fragment (16-bit A 16x32 layout): lane holds K = k0+hi*8+[0..7] and +16
        const int ka = k0 + hi * 8;
        v8h a_lo = *(const v8h*)(arow + ka);
        v8h a_hi = *(const v8h*)(arow + ka + 16);
        v16h a = __builtin_shufflevector(a_lo, a_hi, 0,1,2,3,4,5,6,7,8,9,10,11,12,13,14,15);
        const int kb = k0 + hi * 8;            // B: lane col = m, K = kb+[0..7] and +16
#pragma unroll
        for (int n = 0; n < NTILES; ++n) {
            const _Float16* bp = sB + (size_t)(n * 16 + m) * HP + kb;
            v8h b_lo = *(const v8h*)(bp);
            v8h b_hi = *(const v8h*)(bp + 16);
            v16h bfrag = __builtin_shufflevector(b_lo, b_hi,
                           0,1,2,3,4,5,6,7,8,9,10,11,12,13,14,15);
            acc[n] = __builtin_amdgcn_wmma_f32_16x16x32_f16(false, a, false, bfrag,
                                                            (short)0, acc[n], false, false);
        }
    }
    // C/D layout: VGPR r -> row = r + 8*hi, col = m
#pragma unroll
    for (int n = 0; n < NTILES; ++n) {
        float* crow = C + (size_t)(tileM * 16 + hi * 8) * HP + n * 16 + m;
#pragma unroll
        for (int r = 0; r < 8; ++r) crow[(size_t)r * HP] = acc[n][r];
    }
}

// ---------------- edge scatter: out[col] += hW[row] * norm ----------------
__global__ void k_scatter(const float* __restrict__ hW, const int* __restrict__ row,
                          const int* __restrict__ col, const float* __restrict__ norm,
                          float* __restrict__ out) {
    int e = blockIdx.x;
    int f = threadIdx.x;
    if (f >= H) return;
    float nm = norm[e];
    atomicAdd(&out[(size_t)col[e] * HP + f], hW[(size_t)row[e] * HP + f] * nm);
}

// ---------------- self-loop contribution (layer 0 only) ----------------
__global__ void k_selfloop(const float* __restrict__ hW, const float* __restrict__ selfc,
                           float* __restrict__ out) {
    int i = blockIdx.x;
    int f = threadIdx.x;
    if (f < H) out[(size_t)i * HP + f] += hW[(size_t)i * HP + f] * selfc[i];
}

// ---------------- bias (+ optional relu); dual-writes f32 + f16 for next GEMM ----------
__global__ void k_bias_act(float* __restrict__ h, _Float16* __restrict__ h16,
                           const float* __restrict__ b, int do_relu) {
    int i = blockIdx.x;
    int f = threadIdx.x;   // 0..HP-1
    float v = 0.f;
    if (f < H) {
        v = h[(size_t)i * HP + f] + b[f];
        if (do_relu) v = fmaxf(v, 0.f);
    }
    h[(size_t)i * HP + f]   = v;
    h16[(size_t)i * HP + f] = (_Float16)v;
}

// ---------------- global mean pool ----------------
__global__ void k_cnt(const int* __restrict__ batch, float* cnt) {
    int i = blockIdx.x * blockDim.x + threadIdx.x;
    if (i < N_NODES) atomicAdd(&cnt[batch[i]], 1.f);
}
__global__ void k_pool(const float* __restrict__ h, const int* __restrict__ batch,
                       float* __restrict__ pooled) {
    int i = blockIdx.x;
    int f = threadIdx.x;
    if (f < H) atomicAdd(&pooled[(size_t)batch[i] * H + f], h[(size_t)i * HP + f]);
}
__global__ void k_pool_div(float* pooled, const float* cnt) {
    int idx = blockIdx.x * blockDim.x + threadIdx.x;
    if (idx >= GRAPHS * H) return;
    int g = idx / H;
    pooled[idx] /= fmaxf(cnt[g], 1.f);
}

// ---------------- head: train-mode BN (biased var) + ReLU ----------------
__global__ void k_bn_relu(const float* __restrict__ in, float* __restrict__ out,
                          const float* __restrict__ gamma, const float* __restrict__ beta,
                          int rows, int cols) {
    int f = blockIdx.x * blockDim.x + threadIdx.x;
    if (f >= cols) return;
    float mu = 0.f;
    for (int r = 0; r < rows; ++r) mu += in[r * cols + f];
    mu /= (float)rows;
    float var = 0.f;
    for (int r = 0; r < rows; ++r) { float d = in[r * cols + f] - mu; var += d * d; }
    var /= (float)rows;
    float s = rsqrtf(var + 1e-5f) * gamma[f];
    float bb = beta[f];
    for (int r = 0; r < rows; ++r) {
        float v = (in[r * cols + f] - mu) * s + bb;
        out[r * cols + f] = fmaxf(v, 0.f);
    }
}

// ---------------- head: small dense linear ----------------
__global__ void k_linear(const float* __restrict__ in, const float* __restrict__ W,
                         const float* __restrict__ b, float* __restrict__ out,
                         int rows, int K, int Nout) {
    int idx = blockIdx.x * blockDim.x + threadIdx.x;
    if (idx >= rows * Nout) return;
    int r = idx / Nout, j = idx % Nout;
    float acc = b[j];
    for (int k = 0; k < K; ++k) acc += in[r * K + k] * W[k * Nout + j];
    out[r * Nout + j] = acc;
}

extern "C" void kernel_launch(void* const* d_in, const int* in_sizes, int n_in,
                              void* d_out, int out_size, void* d_ws, size_t ws_size,
                              hipStream_t stream) {
    (void)in_sizes; (void)n_in; (void)out_size; (void)ws_size;
    const float* x     = (const float*)d_in[0];
    const float* ew    = (const float*)d_in[1];
    const float* W_in  = (const float*)d_in[2];
    const float* b_in  = (const float*)d_in[3];
    const float* convW[4] = {(const float*)d_in[4], (const float*)d_in[5],
                             (const float*)d_in[6], (const float*)d_in[7]};
    const float* convB[4] = {(const float*)d_in[8], (const float*)d_in[9],
                             (const float*)d_in[10], (const float*)d_in[11]};
    const float* bn1_g = (const float*)d_in[12];
    const float* bn1_b = (const float*)d_in[13];
    const float* W1    = (const float*)d_in[14];
    const float* b1    = (const float*)d_in[15];
    const float* bn2_g = (const float*)d_in[16];
    const float* bn2_b = (const float*)d_in[17];
    const float* W2    = (const float*)d_in[18];
    const float* b2    = (const float*)d_in[19];
    const float* bn3_g = (const float*)d_in[20];
    const float* bn3_b = (const float*)d_in[21];
    const float* W3    = (const float*)d_in[22];
    const float* b3    = (const float*)d_in[23];
    const int*   eidx  = (const int*)d_in[24];
    const int*   batch = (const int*)d_in[25];
    const int* erow = eidx;
    const int* ecol = eidx + N_EDGES;

    // ---- carve workspace ----
    char* ws = (char*)d_ws;
    size_t off = 0;
    auto carve = [&](size_t bytes) -> void* {
        void* p = ws + off;
        off = (off + bytes + 255) & ~(size_t)255;
        return p;
    };
    float*    hA     = (float*)   carve((size_t)N_NODES * HP * 4);
    float*    hB     = (float*)   carve((size_t)N_NODES * HP * 4);
    _Float16* hf16   = (_Float16*)carve((size_t)N_NODES * HP * 2);
    _Float16* Wt     = (_Float16*)carve((size_t)HP * HP * 2);
    float*    deg    = (float*)   carve((size_t)N_NODES * 4);
    float*    dinv   = (float*)   carve((size_t)N_NODES * 4);
    float*    dinv0  = (float*)   carve((size_t)N_NODES * 4);
    float*    selfc  = (float*)   carve((size_t)N_NODES * 4);
    float*    normA  = (float*)   carve((size_t)N_EDGES * 4);
    float*    normB  = (float*)   carve((size_t)N_EDGES * 4);
    float*    pooled = (float*)   carve((size_t)GRAPHS * H * 4);
    float*    cnt    = (float*)   carve((size_t)GRAPHS * 4);
    float*    gb1    = (float*)   carve((size_t)GRAPHS * H * 4);
    float*    t1     = (float*)   carve((size_t)GRAPHS * H2 * 4);
    float*    gb2    = (float*)   carve((size_t)GRAPHS * H2 * 4);
    float*    t2     = (float*)   carve((size_t)GRAPHS * H4 * 4);
    float*    gb3    = (float*)   carve((size_t)GRAPHS * H4 * 4);

    const int NHP = N_NODES * HP;

    // ---- input linear into hA (+ f16 staging) ----
    k_input_linear<<<N_NODES, HP, 0, stream>>>(x, W_in, b_in, hA, hf16);

    // ---- degree / norm precompute ----
    k_fill<<<(N_NODES + 255) / 256, 256, 0, stream>>>(deg, 0.f, N_NODES);
    k_deg<<<(N_EDGES + 255) / 256, 256, 0, stream>>>(ecol, ew, deg);
    k_dinv<<<(N_NODES + 255) / 256, 256, 0, stream>>>(deg, dinv, dinv0, selfc);
    k_norm<<<(N_EDGES + 255) / 256, 256, 0, stream>>>(erow, ecol, ew, dinv, dinv0, normB, normA);

    // ---- 4 GCN layers ----
    float* h_cur = hA;
    float* h_oth = hB;
    const int gemm_blocks = (MTILES + WAVES_PER_BLOCK - 1) / WAVES_PER_BLOCK;   // 391
    for (int l = 0; l < 4; ++l) {
        k_w_to_f16t<<<(HP * HP + 255) / 256, 256, 0, stream>>>(convW[l], Wt);
        k_wmma_gemm<<<gemm_blocks, 256, 0, stream>>>(hf16, Wt, h_cur);   // h_cur := hW
        k_fill<<<(NHP + 255) / 256, 256, 0, stream>>>(h_oth, 0.f, NHP);
        k_scatter<<<N_EDGES, HP, 0, stream>>>(h_cur, erow, ecol,
                                              (l == 0) ? normA : normB, h_oth);
        if (l == 0) k_selfloop<<<N_NODES, HP, 0, stream>>>(h_cur, selfc, h_oth);
        k_bias_act<<<N_NODES, HP, 0, stream>>>(h_oth, hf16, convB[l], (l < 3) ? 1 : 0);
        float* tmp = h_cur; h_cur = h_oth; h_oth = tmp;
    }

    // ---- global mean pool ----
    k_fill<<<(GRAPHS * H + 255) / 256, 256, 0, stream>>>(pooled, 0.f, GRAPHS * H);
    k_fill<<<1, 256, 0, stream>>>(cnt, 0.f, GRAPHS);
    k_cnt<<<(N_NODES + 255) / 256, 256, 0, stream>>>(batch, cnt);
    k_pool<<<N_NODES, HP, 0, stream>>>(h_cur, batch, pooled);
    k_pool_div<<<(GRAPHS * H + 255) / 256, 256, 0, stream>>>(pooled, cnt);

    // ---- head ----
    k_bn_relu<<<(H + 127) / 128, 128, 0, stream>>>(pooled, gb1, bn1_g, bn1_b, GRAPHS, H);
    k_linear<<<(GRAPHS * H2 + 255) / 256, 256, 0, stream>>>(gb1, W1, b1, t1, GRAPHS, H, H2);
    k_bn_relu<<<1, 128, 0, stream>>>(t1, gb2, bn2_g, bn2_b, GRAPHS, H2);
    k_linear<<<(GRAPHS * H4 + 255) / 256, 256, 0, stream>>>(gb2, W2, b2, t2, GRAPHS, H2, H4);
    k_bn_relu<<<1, 64, 0, stream>>>(t2, gb3, bn3_g, bn3_b, GRAPHS, H4);
    k_linear<<<(GRAPHS * NCLS + 255) / 256, 256, 0, stream>>>(gb3, W3, b3, (float*)d_out,
                                                             GRAPHS, H4, NCLS);
}
